// BoxSampler_32719060861126
// MI455X (gfx1250) — matrix operand order repaired and checked by reading, not compiled, not run on hardware
//
#include <hip/hip_runtime.h>
#include <stdint.h>

// ---------------------------------------------------------------------------
// BalancedPositiveNegativeSampler for MI455X (gfx1250, wave32).
//
// Memory-bound (24 MB in / 128 KB out => ~1.1 us at 23.3 TB/s). Core work is
// prefix-sum / stream compaction:
//   * Block scans via V_WMMA_F32_16X16X32_F16: excl-scan(x) = x * T with
//     T[k][n] = (k < n). The spare M dimension of the 16x16 accumulator
//     carries a SECOND vector (rows m<8 / m>=8), so one WMMA pair scans the
//     positive-count and negative-count vectors simultaneously. Counts are
//     <= 32 (wave32 ballot popcounts): exact in f16, exact in f32 acc.
//   * Tile streaming via GLOBAL_LOAD_ASYNC_TO_LDS_B32 (double buffered),
//     ordered with s_wait_asynccnt: async loads complete in order per wave,
//     so waiting ASYNCcnt<=3 after issuing tile t+1 guarantees tile t landed.
// ---------------------------------------------------------------------------

typedef __attribute__((ext_vector_type(16))) _Float16 v16h;
typedef __attribute__((ext_vector_type(8)))  float    v8f;

#define BATCH   64
#define NLEN    131072
#define NSAMP   512
#define MAXPOS  128          // NUM_SAMPLES * FG_FRAC
#define BLK     1024
#define NWAVE   32           // 1024 / wave32
#define NTILE   (NLEN / BLK) // 128 tiles per row
#define NWORDS  (NLEN / 32)  // 4096 selection-mask words per row

__device__ __forceinline__ unsigned ballot32(bool p) {
  return __builtin_amdgcn_ballot_w32(p);
}

// Sum of 4 bytes (each 0/1) packed in a dword.
__device__ __forceinline__ int sum4(uint32_t w) {
  return (int)((w * 0x01010101u) >> 24);
}

// Async copy of one dword from global (SADDR base + 32-bit voff, GVS mode)
// into LDS at the byte offset of lds_dst.  Tracked by ASYNCcnt.
__device__ __forceinline__ void async_b32(void* lds_dst,
                                          const void* sbase,
                                          uint32_t voff) {
  asm volatile("global_load_async_to_lds_b32 %0, %1, %2"
               :
               : "v"((uint32_t)(uintptr_t)lds_dst), "v"(voff),
                 "s"((unsigned long long)(uintptr_t)sbase)
               : "memory");
}

// Dual exclusive prefix scan of two 32-element vectors xa, xb via one WMMA
// pair.  Must be executed by full wave(s) (EXEC all-ones).
//   A[m][k] = (m < 8) ? xa[k] : xb[k]              (16x32 f16, ISA A-layout)
//   Blo[k][n] = (k < n),  Bhi[k][n] = (k < n+16)   (32x16 f16, ISA B-layout)
//   D[m][n]  = sum_{k<n(+16)} x[k]                 (16x16 f32, ISA C/D layout)
// Extraction (vgpr 0 of D): lanes 0-15 -> M=0 (xa), N=lane;
//                           lanes 16-31 -> M=8 (xb), N=lane-16.
__device__ __forceinline__ void wmma_dual_scan32(const float* xa,
                                                 const float* xb,
                                                 int lane,
                                                 int* preA, int* preB) {
  const int g = lane >> 4;      // K-half selector for A / B layouts
  const int m = lane & 15;      // A-matrix row
  const int n = lane & 15;      // B-matrix column
  v16h A, Blo, Bhi;
#pragma unroll
  for (int j = 0; j < 16; ++j) {
    // A in contiguous k order: element j of lane half g -> K = j + 16g (matches B)
    const int ka = j + 16 * g;
    A[j] = (_Float16)((m < 8) ? xa[ka] : xb[ka]);
    // B in contiguous k order: element j of lane half g -> row k = j + 16g, column n = lane%16
    const int kb = j + 16 * g;
    Blo[j] = (_Float16)((kb < n)      ? 1.0f : 0.0f);
    Bhi[j] = (_Float16)((kb < n + 16) ? 1.0f : 0.0f);
  }
  v8f c0 = {};
  v8f c1 = {};
  c0 = __builtin_amdgcn_wmma_f32_16x16x32_f16(false, A, false, Blo, (short)0,
                                              c0, false, false);
  c1 = __builtin_amdgcn_wmma_f32_16x16x32_f16(false, A, false, Bhi, (short)0,
                                              c1, false, false);
  if (lane < 16) {
    preA[n]      = (int)c0[0];   // M=0 row: exclusive prefix of xa
    preA[16 + n] = (int)c1[0];
  } else {
    preB[n]      = (int)c0[0];   // M=8 row: exclusive prefix of xb
    preB[16 + n] = (int)c1[0];
  }
}

// ---------------------------------------------------------------------------
// Kernel 1: per-row balanced selection -> 1-bit-per-element mask in d_ws.
// One 1024-thread block per batch row.
// ---------------------------------------------------------------------------
__global__ __launch_bounds__(BLK) void box_select_kernel(
    const uint8_t* __restrict__ pos, const uint8_t* __restrict__ neg,
    const uint8_t* __restrict__ ign, uint32_t* __restrict__ selmask) {
  const int row  = blockIdx.x;
  const int tid  = threadIdx.x;
  const int lane = tid & 31;
  const int wid  = tid >> 5;
  const unsigned lt_mask = (lane == 0) ? 0u : (0xFFFFFFFFu >> (32 - lane));

  __shared__ int     s_cnt;
  __shared__ float   s_posc[NWAVE], s_negc[NWAVE];
  __shared__ int     s_posPre[NWAVE], s_negPre[NWAVE];
  __shared__ int     s_carry[2];
  __shared__ uint8_t s_buf[2][3][BLK];   // double-buffered {pos,neg,ign} tiles

  const uint8_t* prow = pos + (size_t)row * NLEN;
  const uint8_t* nrow = neg + (size_t)row * NLEN;
  const uint8_t* irow = ign + (size_t)row * NLEN;
  uint32_t* mrow = selmask + (size_t)row * NWORDS;

  // ---- Phase 1: count positives among candidates (pos & ~ign) -------------
  if (tid == 0) s_cnt = 0;
  __syncthreads();
  {
    const uint4* p4 = (const uint4*)prow;
    const uint4* i4 = (const uint4*)irow;
    int c = 0;
    for (int i = tid; i < NLEN / 16; i += BLK) {
      __builtin_prefetch(&p4[i + BLK], 0, 0);   // global_prefetch_b8
      uint4 pv = p4[i];
      uint4 iv = i4[i];
      c += sum4(pv.x & ~iv.x) + sum4(pv.y & ~iv.y) +
           sum4(pv.z & ~iv.z) + sum4(pv.w & ~iv.w);
    }
    atomicAdd(&s_cnt, c);   // integer add: order-independent, deterministic
  }
  __syncthreads();
  const int num_pos = min(MAXPOS, s_cnt);
  const int num_neg = NSAMP - num_pos;

  // ---- Phase 2: permuted-tile traversal, take first-k pos / first-k neg ---
  if (tid == 0) { s_carry[0] = 0; s_carry[1] = 0; }

  // Prologue: async-stage tile 0 into buffer 0 (threads 0..255, 1 dword each).
  if (tid < BLK / 4) {
    const int tp0 = (row * 19) & (NTILE - 1);           // tile_of(0)
    const uint32_t voff = (uint32_t)(tp0 * BLK + tid * 4);
    async_b32(&s_buf[0][0][tid * 4], prow, voff);
    async_b32(&s_buf[0][1][tid * 4], nrow, voff);
    async_b32(&s_buf[0][2][tid * 4], irow, voff);
  }
  __syncthreads();

  for (int t = 0; t < NTILE; ++t) {
    const int cur = t & 1;

    // Stage tile t+1 into the other buffer while we process tile t.
    if (t + 1 < NTILE) {
      if (tid < BLK / 4) {
        const int tpn = ((t + 1) * 45 + row * 19) & (NTILE - 1);
        const uint32_t voff = (uint32_t)(tpn * BLK + tid * 4);
        async_b32(&s_buf[cur ^ 1][0][tid * 4], prow, voff);
        async_b32(&s_buf[cur ^ 1][1][tid * 4], nrow, voff);
        async_b32(&s_buf[cur ^ 1][2][tid * 4], irow, voff);
      }
      // Async loads complete in order: <=3 outstanding => tile t has landed.
      asm volatile("s_wait_asynccnt 3" ::: "memory");
    } else {
      asm volatile("s_wait_asynccnt 0" ::: "memory");
    }
    __syncthreads();   // tile t visible to all waves

    const int tp = (t * 45 + row * 19) & (NTILE - 1);   // bijective tile order
    const int i  = tp * BLK + tid;
    const uint8_t pv = s_buf[cur][0][tid];
    const uint8_t nv = s_buf[cur][1][tid];
    const uint8_t iv = s_buf[cur][2][tid];
    const bool isP = (pv != 0) && (iv == 0);
    const bool isN = (nv != 0) && (iv == 0) && !isP;

    const unsigned pb = ballot32(isP);
    const unsigned nb = ballot32(isN);
    if (lane == 0) {
      s_posc[wid] = (float)__popc(pb);
      s_negc[wid] = (float)__popc(nb);
    }
    __syncthreads();

    if (wid == 0)   // one full wave, uniform branch -> EXEC all-ones
      wmma_dual_scan32(s_posc, s_negc, lane, s_posPre, s_negPre);
    __syncthreads();

    const int posBefore = s_carry[0] + s_posPre[wid] + __popc(pb & lt_mask);
    const int negBefore = s_carry[1] + s_negPre[wid] + __popc(nb & lt_mask);
    const bool sel = (isP && posBefore < num_pos) ||
                     (isN && negBefore < num_neg);
    const unsigned sb = ballot32(sel);
    if (lane == 0) mrow[(unsigned)i >> 5] = sb;  // one word per (tile, wave)
    __syncthreads();

    if (tid == 0) {
      s_carry[0] += s_posPre[NWAVE - 1] + (int)s_posc[NWAVE - 1];
      s_carry[1] += s_negPre[NWAVE - 1] + (int)s_negc[NWAVE - 1];
    }
    __syncthreads();
  }
}

// ---------------------------------------------------------------------------
// Kernel 2: ascending-index compaction of the selection mask -> int32 indices.
// Matches reference top_k ordering: selected indices ascending.
// ---------------------------------------------------------------------------
__global__ __launch_bounds__(BLK) void box_compact_kernel(
    const uint32_t* __restrict__ selmask, int* __restrict__ out) {
  const int row  = blockIdx.x;
  const int tid  = threadIdx.x;
  const int lane = tid & 31;
  const int wid  = tid >> 5;

  __shared__ float s_cnt[BLK];
  __shared__ int   s_lanePre[BLK];
  __shared__ int   s_dummy[32];    // sink for the unused xb scan (races OK)
  __shared__ int   s_wpre[NWAVE];
  __shared__ int   s_carry;

  const uint32_t* mrow = selmask + (size_t)row * NWORDS;
  int* orow = out + row * NSAMP;

  if (tid < NSAMP) orow[tid] = 0;   // default for (degenerate) unfilled slots
  if (tid == 0) s_carry = 0;
  __syncthreads();

  for (int c = 0; c < NWORDS / BLK; ++c) {   // 4 chunks of 1024 words
    const int widx = c * BLK + tid;
    const uint32_t w = mrow[widx];
    s_cnt[tid] = (float)__popc(w);
    __syncthreads();

    // 32 concurrent intra-wave exclusive scans, one WMMA pair per wave.
    wmma_dual_scan32(&s_cnt[wid * 32], &s_cnt[wid * 32], lane,
                     &s_lanePre[wid * 32], s_dummy);
    __syncthreads();

    if (tid == 0) {   // serial 32-entry cross-wave scan (negligible)
      int run = s_carry;
#pragma unroll
      for (int ww = 0; ww < NWAVE; ++ww) {
        s_wpre[ww] = run;
        run += s_lanePre[ww * 32 + 31] + (int)s_cnt[ww * 32 + 31];
      }
      s_carry = run;
    }
    __syncthreads();

    int p = s_wpre[wid] + s_lanePre[tid];
    uint32_t bits = w;
    while (bits) {
      const int b = __ffs(bits) - 1;
      bits &= bits - 1;
      if (p < NSAMP) orow[p] = widx * 32 + b;
      ++p;
    }
    __syncthreads();
  }
}

// ---------------------------------------------------------------------------
extern "C" void kernel_launch(void* const* d_in, const int* in_sizes, int n_in,
                              void* d_out, int out_size, void* d_ws,
                              size_t ws_size, hipStream_t stream) {
  const uint8_t* pos = (const uint8_t*)d_in[0];  // positive_matches [64,131072] bool
  const uint8_t* neg = (const uint8_t*)d_in[1];  // negative_matches
  const uint8_t* ign = (const uint8_t*)d_in[2];  // ignored_matches
  uint32_t* selmask  = (uint32_t*)d_ws;          // 64 * 4096 words = 1 MB
  int* out           = (int*)d_out;              // int32 [64, 512]
  (void)in_sizes; (void)n_in; (void)out_size; (void)ws_size;

  hipLaunchKernelGGL(box_select_kernel, dim3(BATCH), dim3(BLK), 0, stream,
                     pos, neg, ign, selmask);
  hipLaunchKernelGGL(box_compact_kernel, dim3(BATCH), dim3(BLK), 0, stream,
                     selmask, out);
}
